// SelfAttentionLayer_83382495085170
// MI455X (gfx1250) — matrix operand order repaired and checked by reading, not compiled, not run on hardware
//
#include <hip/hip_runtime.h>
#include <math.h>

// Problem constants (match reference)
#define BB   4
#define SS   2048
#define EE   2048
#define HH   16
#define DD   128
#define MTOT (BB*SS)      // 8192 rows of x
#define LDT  40           // padded LDS halfword stride (80B, multiple of 16B)

typedef __attribute__((ext_vector_type(16))) __bf16 v16bf;
typedef __attribute__((ext_vector_type(8)))  float  v8f;

// Load a 16-element bf16 WMMA fragment: 8 contiguous halfwords at p, 8 more at p+16.
// Matches the CDNA5 16-bit A/B VGPR layout: lane<16 holds K{0..7,16..23} of its
// row/col, lane>=16 holds K{8..15,24..31} (caller adds half*8 to the base).
__device__ __forceinline__ v16bf frag16(const __bf16* p) {
  union { v16bf v; float4 f[2]; } u;
  u.f[0] = *reinterpret_cast<const float4*>(p);
  u.f[1] = *reinterpret_cast<const float4*>(p + 16);
  return u.v;
}

// ---------------------------------------------------------------------------
// fp32 -> bf16 elementwise cast (x)
// ---------------------------------------------------------------------------
__global__ void cast_x_kernel(const float* __restrict__ x, __bf16* __restrict__ xb, int n4) {
  int i = blockIdx.x * blockDim.x + threadIdx.x;
  if (i < n4) {
    float4 f = reinterpret_cast<const float4*>(x)[i];
    union { __bf16 h[4]; uint2 u; } p;
    p.h[0] = (__bf16)f.x; p.h[1] = (__bf16)f.y;
    p.h[2] = (__bf16)f.z; p.h[3] = (__bf16)f.w;
    reinterpret_cast<uint2*>(xb)[i] = p.u;
  }
}

// ---------------------------------------------------------------------------
// W [K][N] fp32 -> W^T [N][K] bf16, LDS-tiled 32x32 transpose
// ---------------------------------------------------------------------------
__global__ __launch_bounds__(256)
void transpose_w_kernel(const float* __restrict__ W, __bf16* __restrict__ WT) {
  __shared__ __bf16 t[32][33];
  const int n0 = blockIdx.x * 32, k0 = blockIdx.y * 32;
  const int tx = threadIdx.x, ty = threadIdx.y;   // block (32,8)
#pragma unroll
  for (int i = 0; i < 4; i++)
    t[ty + 8*i][tx] = (__bf16)W[(size_t)(k0 + ty + 8*i) * EE + n0 + tx];
  __syncthreads();
#pragma unroll
  for (int i = 0; i < 4; i++)
    WT[(size_t)(n0 + ty + 8*i) * EE + k0 + tx] = t[tx][ty + 8*i];
}

// ---------------------------------------------------------------------------
// bf16 GEMM: C[M,N] = A[M,K] * (B^T)[N,K]^T + bias, f32 accumulate via WMMA.
// Block = 256 threads (8 waves), 128(M) x 64(N) tile; wave = 32x32 (2x2 WMMA).
// Tiles staged with GLOBAL_LOAD_ASYNC_TO_LDS_B128 (ASYNCcnt), double-buffered:
// chunk k+1's async DMA overlaps chunk k's WMMA stream; per-wave
// s_wait_asynccnt 0x3 gates only on the current buffer's 3 outstanding ops.
// Epilogue MODE (compile-time):
//   0: Q/K -> bf16 [B][H][S][D]  (scale folded; pass 1.0 for K)
//   2: V   -> bf16 [B][H][D][S]  (transposed for the P*V B-fragment)
//   3: O   -> fp32 [M][N]        (pre-residual main_out)
// ---------------------------------------------------------------------------
template <int MODE>
__global__ __launch_bounds__(256)
void gemm_bf16_kernel(const __bf16* __restrict__ A,   // [M][K] row-major
                      const __bf16* __restrict__ BT,  // [N][K] row-major (= W^T)
                      const float*  __restrict__ bias,
                      void* __restrict__ Out,
                      int Kdim, int Ndim, float scale) {
  __shared__ __bf16 As[2][128 * LDT];
  __shared__ __bf16 Bs[2][64 * LDT];

  const int tid  = threadIdx.x;
  const int lane = tid & 31, wid = tid >> 5;
  const int lq   = lane & 15, half = lane >> 4;
  const int wm   = wid & 3,  wn  = wid >> 2;       // 4 m-waves x 2 n-waves
  const int m0   = blockIdx.y * 128;
  const int n0   = blockIdx.x * 64;

  const v8f vzero = {0,0,0,0,0,0,0,0};
  v8f acc[2][2];
#pragma unroll
  for (int mt = 0; mt < 2; mt++)
#pragma unroll
    for (int nt = 0; nt < 2; nt++) acc[mt][nt] = vzero;

  const int arow = tid >> 1;             // 128 rows, 2 threads/row, 32B each
  const int acg  = (tid & 1) * 16;
  const int brow = tid >> 2;             // 64 rows, 4 threads/row, 16B each
  const int bcg  = (tid & 3) * 8;

  const __bf16* ga = A  + (size_t)(m0 + arow) * Kdim + acg;
  const __bf16* gb = BT + (size_t)(n0 + brow) * Kdim + bcg;

  // Per-thread LDS byte addresses (low 32 bits of the shared-aperture pointer)
  const unsigned ldsA[2] = { (unsigned)(uintptr_t)&As[0][arow * LDT + acg],
                             (unsigned)(uintptr_t)&As[1][arow * LDT + acg] };
  const unsigned ldsB[2] = { (unsigned)(uintptr_t)&Bs[0][brow * LDT + bcg],
                             (unsigned)(uintptr_t)&Bs[1][brow * LDT + bcg] };

  // 3 async b128 ops per thread per chunk: A 32B (two b128, inst-offset applies
  // to both global and LDS sides), B 16B (one b128).
  auto issue = [&](int kc, int buf) {
    const __bf16* pa = ga + kc;
    const __bf16* pb = gb + kc;
    asm volatile("global_load_async_to_lds_b128 %0, %1, off"
                 :: "v"(ldsA[buf]), "v"(pa) : "memory");
    asm volatile("global_load_async_to_lds_b128 %0, %1, off offset:16"
                 :: "v"(ldsA[buf]), "v"(pa) : "memory");
    asm volatile("global_load_async_to_lds_b128 %0, %1, off"
                 :: "v"(ldsB[buf]), "v"(pb) : "memory");
  };

  auto compute = [&](int buf) {
    v16bf af[2], bf[2];
#pragma unroll
    for (int mt = 0; mt < 2; mt++)
      af[mt] = frag16(&As[buf][(wm*32 + mt*16 + lq) * LDT + half*8]);
#pragma unroll
    for (int nt = 0; nt < 2; nt++)
      bf[nt] = frag16(&Bs[buf][(wn*32 + nt*16 + lq) * LDT + half*8]);
#pragma unroll
    for (int mt = 0; mt < 2; mt++)
#pragma unroll
      for (int nt = 0; nt < 2; nt++)
        acc[mt][nt] = __builtin_amdgcn_wmma_f32_16x16x32_bf16(
            false, af[mt], false, bf[nt], (short)0, acc[mt][nt], false, false);
  };

  issue(0, 0);
  int buf = 0, kc = 0;
  for (; kc + 32 < Kdim; kc += 32, buf ^= 1) {
    if (kc + 64 < Kdim) {                // L2 prefetch two chunks ahead
      __builtin_prefetch(ga + kc + 64, 0, 1);
      __builtin_prefetch(gb + kc + 64, 0, 1);
    }
    issue(kc + 32, buf ^ 1);             // DMA next chunk into other buffer
    asm volatile("s_wait_asynccnt 0x3" ::: "memory");  // current buffer done
    __syncthreads();                     // ... across all waves
    compute(buf);
    __syncthreads();                     // readers done before next overwrite
  }
  asm volatile("s_wait_asynccnt 0x0" ::: "memory");
  __syncthreads();
  compute(buf);                          // tail chunk

  // Epilogue: C layout -> lane holds col (nt*16+lq), rows j + 8*half
#pragma unroll
  for (int mt = 0; mt < 2; mt++) {
    const int mloc = wm*32 + mt*16 + 8*half;
#pragma unroll
    for (int nt = 0; nt < 2; nt++) {
      const int ncol = n0 + wn*32 + nt*16 + lq;
      const float bv = bias[ncol];
#pragma unroll
      for (int j = 0; j < 8; j++) {
        const int m = m0 + mloc + j;
        float v = acc[mt][nt][j] + bv;
        if (MODE == 3) {
          reinterpret_cast<float*>(Out)[(size_t)m * Ndim + ncol] = v;
        } else {
          v *= scale;
          const int b = m >> 11,  s = m & (SS - 1);     // SS = 2048
          const int h = ncol >> 7, d = ncol & (DD - 1); // DD = 128
          __bf16* o = reinterpret_cast<__bf16*>(Out);
          if (MODE == 2)   // V^T: [B][H][D][S]
            o[(((size_t)b*HH + h)*DD + d)*SS + s] = (__bf16)v;
          else             // Q/K: [B][H][S][D]
            o[(((size_t)b*HH + h)*SS + s)*DD + d] = (__bf16)v;
        }
      }
    }
  }
}

// ---------------------------------------------------------------------------
// Flash-style attention: one wave per (b, h, 16-row Q tile).
// Streams keys in chunks of 32; scores + P*V via WMMA bf16; online softmax.
// P C-layout -> A-fragment relayout goes through LDS (in-order DS in a wave).
// ---------------------------------------------------------------------------
__global__ __launch_bounds__(32)
__attribute__((amdgpu_waves_per_eu(1)))
void attn_kernel(const __bf16* __restrict__ Qb,   // [B][H][S][D], pre-scaled
                 const __bf16* __restrict__ Kb,   // [B][H][S][D]
                 const __bf16* __restrict__ VTb,  // [B][H][D][S]
                 const float*  __restrict__ mask, // [B][1][1][S]
                 __bf16* __restrict__ ctx) {      // [B][S][E]
  __shared__ __bf16 pls[16 * LDT];

  const int lane = threadIdx.x;
  const int lq = lane & 15, half = lane >> 4;
  const int q0 = blockIdx.x * 16;
  const int h  = blockIdx.y, b = blockIdx.z;
  const size_t bh = (size_t)b * HH + h;
  const __bf16* Q  = Qb  + bh * SS * DD;
  const __bf16* K  = Kb  + bh * SS * DD;
  const __bf16* VT = VTb + bh * DD * SS;

  // Q tile as 4 A-fragments (16 x 32 each over D=128)
  v16bf qf[4];
#pragma unroll
  for (int dk = 0; dk < 4; dk++)
    qf[dk] = frag16(&Q[(size_t)(q0 + lq) * DD + dk*32 + half*8]);

  const v8f vzero = {0,0,0,0,0,0,0,0};
  v8f O[8];
#pragma unroll
  for (int t = 0; t < 8; t++) O[t] = vzero;
  float mrow[8], lrow[8];
#pragma unroll
  for (int j = 0; j < 8; j++) { mrow[j] = -INFINITY; lrow[j] = 0.0f; }

  for (int kc = 0; kc < SS; kc += 32) {
    // --- scores: two 16x16 tiles over 32 keys, K-dim = D chunks of 32 ---
    v8f s[2];
#pragma unroll
    for (int st = 0; st < 2; st++) {
      v8f acc = vzero;
#pragma unroll
      for (int dk = 0; dk < 4; dk++) {
        v16bf kf = frag16(&K[(size_t)(kc + st*16 + lq) * DD + dk*32 + half*8]);
        acc = __builtin_amdgcn_wmma_f32_16x16x32_bf16(
            false, qf[dk], false, kf, (short)0, acc, false, false);
      }
      const float mv = -1e9f * mask[(size_t)b * SS + kc + st*16 + lq];
#pragma unroll
      for (int j = 0; j < 8; j++) acc[j] += mv;   // same column for all rows
      s[st] = acc;
    }

    // --- online softmax (row r = j + 8*half lives in 16-lane halves) ---
    float mnew[8], alpha[8];
#pragma unroll
    for (int j = 0; j < 8; j++) {
      float t = fmaxf(s[0][j], s[1][j]);
      t = fmaxf(t, __shfl_xor(t, 1));
      t = fmaxf(t, __shfl_xor(t, 2));
      t = fmaxf(t, __shfl_xor(t, 4));
      t = fmaxf(t, __shfl_xor(t, 8));
      mnew[j]  = fmaxf(mrow[j], t);
      alpha[j] = __expf(mrow[j] - mnew[j]);
      mrow[j]  = mnew[j];
    }
#pragma unroll
    for (int j = 0; j < 8; j++) {
      s[0][j] = __expf(s[0][j] - mnew[j]);
      s[1][j] = __expf(s[1][j] - mnew[j]);
      float t = s[0][j] + s[1][j];
      t += __shfl_xor(t, 1);
      t += __shfl_xor(t, 2);
      t += __shfl_xor(t, 4);
      t += __shfl_xor(t, 8);
      lrow[j] = lrow[j] * alpha[j] + t;
    }
#pragma unroll
    for (int t = 0; t < 8; t++)
#pragma unroll
      for (int j = 0; j < 8; j++)
        O[t][j] *= alpha[j];

    // --- P (C-layout) -> LDS -> A-fragment (16 x 32) ---
#pragma unroll
    for (int st = 0; st < 2; st++)
#pragma unroll
      for (int j = 0; j < 8; j++)
        pls[(j + 8*half) * LDT + st*16 + lq] = (__bf16)s[st][j];
    __syncthreads();
    v16bf pf = frag16(&pls[lq * LDT + half*8]);
    __syncthreads();

    // --- O += P * V  (B-fragment from V^T: contiguous over keys) ---
#pragma unroll
    for (int dt = 0; dt < 8; dt++) {
      v16bf vf = frag16(&VT[(size_t)(dt*16 + lq) * SS + kc + half*8]);
      O[dt] = __builtin_amdgcn_wmma_f32_16x16x32_bf16(
          false, pf, false, vf, (short)0, O[dt], false, false);
    }
  }

  // normalize rows and emit concat layout [B][S][E]
#pragma unroll
  for (int j = 0; j < 8; j++) {
    const float inv = 1.0f / lrow[j];
    const int srow = q0 + j + 8*half;
    const size_t base = ((size_t)b * SS + srow) * EE + (size_t)h * DD;
#pragma unroll
    for (int dt = 0; dt < 8; dt++)
      ctx[base + dt*16 + lq] = (__bf16)(O[dt][j] * inv);
  }
}

// ---------------------------------------------------------------------------
// residual add + LayerNorm over E=2048, fp32 (reads original x)
// ---------------------------------------------------------------------------
__global__ __launch_bounds__(256)
void ln_kernel(const float* __restrict__ mo, const float* __restrict__ x,
               const float* __restrict__ gamma, const float* __restrict__ beta,
               float* __restrict__ out) {
  __shared__ float red[16];
  const int row = blockIdx.x;
  const int tid = threadIdx.x;
  const size_t base = (size_t)row * EE;

  float v[8], sum = 0.0f, sq = 0.0f;
#pragma unroll
  for (int i = 0; i < 8; i++) {
    const int c = tid + 256*i;
    const float t = mo[base + c] + x[base + c];
    v[i] = t; sum += t; sq += t*t;
  }
#pragma unroll
  for (int m = 1; m < 32; m <<= 1) { sum += __shfl_xor(sum, m); sq += __shfl_xor(sq, m); }
  const int wid = tid >> 5, lane = tid & 31;
  if (lane == 0) { red[wid] = sum; red[8 + wid] = sq; }
  __syncthreads();
  if (tid < 32) {
    float a  = (lane < 8) ? red[lane]     : 0.0f;
    float sb = (lane < 8) ? red[8 + lane] : 0.0f;
#pragma unroll
    for (int m = 1; m < 8; m <<= 1) { a += __shfl_xor(a, m); sb += __shfl_xor(sb, m); }
    if (lane == 0) { red[0] = a; red[1] = sb; }
  }
  __syncthreads();
  sum = red[0]; sq = red[1];
  const float mean = sum * (1.0f / EE);
  const float var  = sq  * (1.0f / EE) - mean * mean;
  const float rs   = rsqrtf(var + 1e-6f);
#pragma unroll
  for (int i = 0; i < 8; i++) {
    const int c = tid + 256*i;
    out[base + c] = (v[i] - mean) * rs * gamma[c] + beta[c];
  }
}

// ---------------------------------------------------------------------------
// host launcher
// Inputs (setup_inputs order): x, mask, Wq, bq, Wk, bk, Wv, bv, Wo, bo, gamma, beta
// Workspace layout (needs 160 MB; regions reused once their producers retire):
//   [  0,  32)MB xb (bf16 x)        -> reused as ctx after QKV GEMMs
//   [ 32,  64)MB WqT,WkT,WvT,WoT (bf16, 8MB each)
//   [ 64, 128)MB Qb, Kb (bf16)      -> reused as fp32 main_out after attention
//   [128, 160)MB VTb (bf16)
// ---------------------------------------------------------------------------
extern "C" void kernel_launch(void* const* d_in, const int* in_sizes, int n_in,
                              void* d_out, int out_size, void* d_ws, size_t ws_size,
                              hipStream_t stream) {
  const float* x    = (const float*)d_in[0];
  const float* mask = (const float*)d_in[1];
  const float* Wq   = (const float*)d_in[2];  const float* bq = (const float*)d_in[3];
  const float* Wk   = (const float*)d_in[4];  const float* bk = (const float*)d_in[5];
  const float* Wv   = (const float*)d_in[6];  const float* bv = (const float*)d_in[7];
  const float* Wo   = (const float*)d_in[8];  const float* bo = (const float*)d_in[9];
  const float* gamma = (const float*)d_in[10];
  const float* beta  = (const float*)d_in[11];
  float* out = (float*)d_out;

  char* ws = (char*)d_ws;
  const size_t MB = 1024u * 1024u;
  __bf16* xb  = (__bf16*)(ws);
  __bf16* WqT = (__bf16*)(ws + 32*MB);
  __bf16* WkT = (__bf16*)(ws + 40*MB);
  __bf16* WvT = (__bf16*)(ws + 48*MB);
  __bf16* WoT = (__bf16*)(ws + 56*MB);
  __bf16* Qb  = (__bf16*)(ws + 64*MB);
  __bf16* Kb  = (__bf16*)(ws + 96*MB);
  __bf16* VTb = (__bf16*)(ws + 128*MB);
  __bf16* ctx = xb;                       // alias: safe after V GEMM completes
  float*  mo  = (float*)(ws + 64*MB);     // alias Qb+Kb: safe after attention

  // 1) casts / weight transposes
  const int n4 = MTOT * EE / 4;
  cast_x_kernel<<<(n4 + 255) / 256, 256, 0, stream>>>(x, xb, n4);
  dim3 tg(EE / 32, EE / 32);
  transpose_w_kernel<<<tg, dim3(32, 8), 0, stream>>>(Wq, WqT);
  transpose_w_kernel<<<tg, dim3(32, 8), 0, stream>>>(Wk, WkT);
  transpose_w_kernel<<<tg, dim3(32, 8), 0, stream>>>(Wv, WvT);
  transpose_w_kernel<<<tg, dim3(32, 8), 0, stream>>>(Wo, WoT);

  // 2) QKV projections (score scale 1/sqrt(E) folded into Q)
  const dim3 gg(EE / 64, MTOT / 128);     // (32, 64)
  const float qscale = 0.022097086912079608f;  // 1/sqrt(2048)
  gemm_bf16_kernel<0><<<gg, 256, 0, stream>>>(xb, WqT, bq, Qb,  EE, EE, qscale);
  gemm_bf16_kernel<0><<<gg, 256, 0, stream>>>(xb, WkT, bk, Kb,  EE, EE, 1.0f);
  gemm_bf16_kernel<2><<<gg, 256, 0, stream>>>(xb, WvT, bv, VTb, EE, EE, 1.0f);

  // 3) flash attention -> ctx (concat heads)
  attn_kernel<<<dim3(SS / 16, HH, BB), 32, 0, stream>>>(Qb, Kb, VTb, mask, ctx);

  // 4) output projection (fp32 out) then residual + LayerNorm
  gemm_bf16_kernel<3><<<gg, 256, 0, stream>>>(ctx, WoT, bo, mo, EE, EE, 1.0f);
  ln_kernel<<<MTOT, 256, 0, stream>>>(mo, x, gamma, beta, out);
}
